// attentionHead_28278064677112
// MI455X (gfx1250) — compile-verified, matrix operand order checked by reading, not run on hardware
//
#include <hip/hip_runtime.h>
#include <hip/hip_bf16.h>

typedef __attribute__((ext_vector_type(16))) _Float16 v16h;
typedef __attribute__((ext_vector_type(8)))  _Float16 h8;
typedef __attribute__((ext_vector_type(8)))  float    v8f;
typedef __attribute__((ext_vector_type(8)))  float    f32x8;

#define HID    768
#define NOUT   2304
#define SEQL   4096
#define NBATCH 2
#define HEADS  12
#define HD     64

// ---------------- shared helpers ----------------

__device__ __forceinline__ v8f v8f_zero() {
  v8f z;
#pragma unroll
  for (int i = 0; i < 8; ++i) z[i] = 0.0f;
  return z;
}

// Load a 16x32 f16 A/B fragment from a row-major LDS tile (stride ld elements).
// ISA 7.12.2 16-bit A 16x32 layout: lanes 0-15 hold K[0..7]+K[16..23] of row M=lane,
// lanes 16-31 hold K[8..15]+K[24..31] of row M=lane-16. B (fed as row-major B^T)
// uses the identical pattern with lane -> N.
__device__ __forceinline__ v16h load_frag(const _Float16* base, int ld, int row0, int k0) {
  const int lane = threadIdx.x & 31;
  const int r    = lane & 15;
  const int kh   = (lane >> 4) & 1;
  const _Float16* p = base + (row0 + r) * ld + k0 + kh * 8;
  h8 lo = *(const h8*)(p);
  h8 hi = *(const h8*)(p + 16);
  v16h f;
#pragma unroll
  for (int i = 0; i < 8; ++i) { f[i] = lo[i]; f[i + 8] = hi[i]; }
  return f;
}

// ---------------- kernel 1: y = x * W^T + b  (f32 in, f16 out), ping-pong LDS ----------------

#define TM 128
#define TN 128
#define TK 64
#define LDT 72   // TK + 8 padding

__global__ __launch_bounds__(256) void qkv_gemm(const float* __restrict__ x,
                                                const float* __restrict__ W,
                                                const float* __restrict__ bias,
                                                _Float16* __restrict__ y) {
  __shared__ _Float16 As[2][TM * LDT];
  __shared__ _Float16 Bs[2][TN * LDT];

  const int tid  = threadIdx.x;
  const int bm   = blockIdx.y * TM;
  const int bn   = blockIdx.x * TN;
  const int lane = tid & 31;
  const int wid  = tid >> 5;        // 0..7
  const int wm   = (wid & 3) * 32;  // 4 waves along M
  const int wn   = (wid >> 2) * 64; // 2 waves along N

  v8f acc[2][4];
#pragma unroll
  for (int i = 0; i < 2; ++i)
#pragma unroll
    for (int j = 0; j < 4; ++j) acc[i][j] = v8f_zero();

  f32x8 ra[4], rw[4];  // global staging registers (one stage ahead)

  auto gload = [&](int k0) {
#pragma unroll
    for (int t = 0; t < 4; ++t) {
      int task = tid + t * 256;          // 0..1023
      int row  = task >> 3;              // 0..127
      int seg  = (task & 7) * 8;         // 0,8,..,56
      ra[t] = *(const f32x8*)(x + (size_t)(bm + row) * HID + k0 + seg);
      rw[t] = *(const f32x8*)(W + (size_t)(bn + row) * HID + k0 + seg);
    }
  };
  auto lstore = [&](int buf) {
#pragma unroll
    for (int t = 0; t < 4; ++t) {
      int task = tid + t * 256;
      int row  = task >> 3;
      int seg  = (task & 7) * 8;
      h8 ah, wh;
#pragma unroll
      for (int i = 0; i < 8; ++i) { ah[i] = (_Float16)ra[t][i]; wh[i] = (_Float16)rw[t][i]; }
      *(h8*)(&As[buf][row * LDT + seg]) = ah;
      *(h8*)(&Bs[buf][row * LDT + seg]) = wh;
    }
  };

  gload(0);
  lstore(0);
  __syncthreads();

  const int NS = HID / TK;  // 12 stages
  for (int s = 0; s < NS; ++s) {
    const int buf = s & 1;
    if (s + 1 < NS) gload((s + 1) * TK);   // issue next-stage global loads early

#pragma unroll
    for (int ks = 0; ks < TK; ks += 32) {
      v16h af[2], bfr[4];
#pragma unroll
      for (int i = 0; i < 2; ++i) af[i] = load_frag(&As[buf][0], LDT, wm + i * 16, ks);
#pragma unroll
      for (int j = 0; j < 4; ++j) bfr[j] = load_frag(&Bs[buf][0], LDT, wn + j * 16, ks);
#pragma unroll
      for (int i = 0; i < 2; ++i)
#pragma unroll
        for (int j = 0; j < 4; ++j)
          acc[i][j] = __builtin_amdgcn_wmma_f32_16x16x32_f16(
              false, af[i], false, bfr[j], (short)0, acc[i][j], false, false);
    }

    if (s + 1 < NS) lstore(buf ^ 1);       // fill the other buffer
    __syncthreads();
  }

  // epilogue: + bias, convert to f16, store row-major (8192 x 2304)
  const int half = lane >> 4;
  const int nl   = lane & 15;
#pragma unroll
  for (int i = 0; i < 2; ++i)
#pragma unroll
    for (int j = 0; j < 4; ++j) {
      const int n    = bn + wn + j * 16 + nl;
      const float bv = bias[n];
#pragma unroll
      for (int r = 0; r < 8; ++r) {
        const int m = bm + wm + i * 16 + r + half * 8;
        y[(size_t)m * NOUT + n] = (_Float16)(acc[i][j][r] + bv);
      }
    }
}

// ---------------- kernel 2: flash attention per (b,h), 128 Q rows / block ----------------

#define QROWS 128
#define KTILE 64
#define LDS2  72

__global__ __launch_bounds__(256) void attn(const _Float16* __restrict__ y,
                                            float* __restrict__ out) {
  __shared__ _Float16 Qs[QROWS * LDS2];
  __shared__ _Float16 Ps[QROWS * LDS2];
  __shared__ _Float16 Ks[2][KTILE * LDS2];  // ping-pong, filled by async DMA
  __shared__ _Float16 Vt[2][HD * LDS2];     // ping-pong, V transposed: [d][key]

  const int tid  = threadIdx.x;
  const int lane = tid & 31;
  const int wid  = tid >> 5;     // 0..7, each wave owns 16 Q rows
  const int half = lane >> 4;
  const int nl   = lane & 15;

  const int qb = blockIdx.x * QROWS;
  const int h  = blockIdx.y;
  const int bz = blockIdx.z;

  // flat-reshape semantics: per batch, head h's Q/K/V are contiguous (4096x64) blocks
  const _Float16* Qg = y + (size_t)bz * ((size_t)SEQL * NOUT) + (size_t)h * (3 * SEQL * HD);
  const _Float16* Kg = Qg + (size_t)SEQL * HD;
  const _Float16* Vg = Kg + (size_t)SEQL * HD;

  // issue async K-tile copy (global -> LDS, no VGPR round trip), 16B per lane
  auto issueK = [&](int k0, int buf) {
#pragma unroll
    for (int c = 0; c < 2; ++c) {
      int task = tid + c * 256;           // 0..511
      int row  = task >> 3;               // 0..63
      int seg  = (task & 7) * 8;
      const _Float16* g = Kg + (size_t)(k0 + row) * HD + seg;
      unsigned l = (unsigned)(size_t)(&Ks[buf][row * LDS2 + seg]);
      asm volatile("global_load_async_to_lds_b128 %0, %1, off"
                   :: "v"(l), "v"(g) : "memory");
    }
  };
  h8 vreg[2];
  auto loadV = [&](int k0) {
#pragma unroll
    for (int c = 0; c < 2; ++c) {
      int task = tid + c * 256;
      int row  = task >> 3;
      int seg  = (task & 7) * 8;
      vreg[c] = *(const h8*)(Vg + (size_t)(k0 + row) * HD + seg);
    }
  };
  auto storeVt = [&](int buf) {
#pragma unroll
    for (int c = 0; c < 2; ++c) {
      int task = tid + c * 256;
      int row  = task >> 3;
      int seg  = (task & 7) * 8;
#pragma unroll
      for (int u = 0; u < 8; ++u) Vt[buf][(seg + u) * LDS2 + row] = vreg[c][u];
    }
  };

  // load Q tile once, pre-scaled by 1/sqrt(64)
#pragma unroll
  for (int t = 0; t < 4; ++t) {
    int task = tid + t * 256;        // 0..1023
    int row  = task >> 3;            // 0..127
    int seg  = (task & 7) * 8;
    h8 qv = *(const h8*)(Qg + (size_t)(qb + row) * HD + seg);
    h8 qs;
#pragma unroll
    for (int i = 0; i < 8; ++i) qs[i] = (_Float16)((float)qv[i] * 0.125f);
    *(h8*)(Qs + row * LDS2 + seg) = qs;
  }

  // stage 0 of K/V
  issueK(0, 0);
  loadV(0);
  storeVt(0);
  asm volatile("s_wait_asynccnt 0x0" ::: "memory");
  __syncthreads();

  v16h qf[2];
  qf[0] = load_frag(Qs, LDS2, wid * 16, 0);
  qf[1] = load_frag(Qs, LDS2, wid * 16, 32);

  float mrow[8], lrow[8];
  v8f accO[4];
#pragma unroll
  for (int r = 0; r < 8; ++r) { mrow[r] = -3.0e38f; lrow[r] = 0.0f; }
#pragma unroll
  for (int j = 0; j < 4; ++j) accO[j] = v8f_zero();

  const int NT = SEQL / KTILE;  // 64 key tiles
  for (int t = 0; t < NT; ++t) {
    const int buf = t & 1;
    if (t + 1 < NT) {                     // prefetch next tile: async K + V regs
      issueK((t + 1) * KTILE, buf ^ 1);
      loadV((t + 1) * KTILE);
    }

    // S = Q * K^T  (16 rows x 64 keys per wave)
    v8f sacc[4];
#pragma unroll
    for (int j = 0; j < 4; ++j) sacc[j] = v8f_zero();
#pragma unroll
    for (int ks = 0; ks < 2; ++ks)
#pragma unroll
      for (int j = 0; j < 4; ++j) {
        v16h kf = load_frag(&Ks[buf][0], LDS2, j * 16, ks * 32);
        sacc[j] = __builtin_amdgcn_wmma_f32_16x16x32_f16(
            false, qf[ks], false, kf, (short)0, sacc[j], false, false);
      }

    // online softmax: row max / rescale / exp / row sum (16-lane half reductions)
    float alpha[8];
#pragma unroll
    for (int r = 0; r < 8; ++r) {
      float v = fmaxf(fmaxf(sacc[0][r], sacc[1][r]), fmaxf(sacc[2][r], sacc[3][r]));
#pragma unroll
      for (int mk = 1; mk < 16; mk <<= 1) v = fmaxf(v, __shfl_xor(v, mk, 32));
      float mnew = fmaxf(mrow[r], v);
      alpha[r] = __expf(mrow[r] - mnew);
      mrow[r]  = mnew;
    }
#pragma unroll
    for (int r = 0; r < 8; ++r) {
      float s = 0.0f;
#pragma unroll
      for (int j = 0; j < 4; ++j) {
        float p = __expf(sacc[j][r] - mrow[r]);
        sacc[j][r] = p;
        s += p;
      }
#pragma unroll
      for (int mk = 1; mk < 16; mk <<= 1) s += __shfl_xor(s, mk, 32);
      lrow[r] = lrow[r] * alpha[r] + s;
#pragma unroll
      for (int j = 0; j < 4; ++j) accO[j][r] *= alpha[r];
    }

    // stage P in LDS: accumulator layout -> A-fragment layout (same-wave rows only)
#pragma unroll
    for (int j = 0; j < 4; ++j)
#pragma unroll
      for (int r = 0; r < 8; ++r)
        Ps[(wid * 16 + r + half * 8) * LDS2 + j * 16 + nl] = (_Float16)sacc[j][r];

    // O += P * V
#pragma unroll
    for (int ks = 0; ks < 2; ++ks) {
      v16h pf = load_frag(Ps, LDS2, wid * 16, ks * 32);
#pragma unroll
      for (int j = 0; j < 4; ++j) {
        v16h vf = load_frag(&Vt[buf][0], LDS2, j * 16, ks * 32);
        accO[j] = __builtin_amdgcn_wmma_f32_16x16x32_f16(
            false, pf, false, vf, (short)0, accO[j], false, false);
      }
    }

    if (t + 1 < NT) storeVt(buf ^ 1);     // fill other V buffer
    asm volatile("s_wait_asynccnt 0x0" ::: "memory");  // next K tile landed
    __syncthreads();
  }

  // epilogue: O / l, write to (B, SEQ, 768) with head transpose folded in
#pragma unroll
  for (int r = 0; r < 8; ++r) {
    const float inv = 1.0f / lrow[r];
    const int q = qb + wid * 16 + r + half * 8;
#pragma unroll
    for (int j = 0; j < 4; ++j) {
      const int d = j * 16 + nl;
      out[((size_t)bz * SEQL + q) * HID + h * HD + d] = accO[j][r] * inv;
    }
  }
}

// ---------------- launch ----------------

extern "C" void kernel_launch(void* const* d_in, const int* in_sizes, int n_in,
                              void* d_out, int out_size, void* d_ws, size_t ws_size,
                              hipStream_t stream) {
  (void)in_sizes; (void)n_in; (void)out_size; (void)ws_size;
  const float* x    = (const float*)d_in[0];
  const float* W    = (const float*)d_in[1];
  const float* bias = (const float*)d_in[2];
  _Float16* y = (_Float16*)d_ws;           // 8192 x 2304 f16 = 37.7 MB scratch
  float* out  = (float*)d_out;

  qkv_gemm<<<dim3(NOUT / TN, (NBATCH * SEQL) / TM), 256, 0, stream>>>(x, W, bias, y);
  attn<<<dim3(SEQL / QROWS, HEADS, NBATCH), 256, 0, stream>>>(y, out);
}
